// RNNLayer_2413771621096
// MI455X (gfx1250) — compile-verified, matrix operand order checked by reading, not run on hardware
//
#include <hip/hip_runtime.h>
#include <hip/hip_bf16.h>

#define B_  64
#define S_  1024
#define D_  512
#define H_  512
#define BS_ (B_ * S_)  // 65536

typedef __bf16 bf16_t;
typedef __attribute__((ext_vector_type(16))) __bf16 v16bf;
typedef __attribute__((ext_vector_type(8)))  __bf16 v8bf;
typedef __attribute__((ext_vector_type(8)))  float  v8f;
typedef __attribute__((ext_vector_type(4)))  float  v4f;

static __device__ __forceinline__ v16bf cat16(v8bf lo, v8bf hi) {
  return __builtin_shufflevector(lo, hi, 0,1,2,3,4,5,6,7,8,9,10,11,12,13,14,15);
}

// ---------------- k0: convert weights f32 -> bf16 (ws) ----------------
// layout: [Wi_f | Wi_b | Wh_f | Wh_b], each H_*D_ = 262144 bf16
__global__ void convert_weights(const float* __restrict__ wi_f,
                                const float* __restrict__ wi_b,
                                const float* __restrict__ wh_f,
                                const float* __restrict__ wh_b,
                                bf16_t* __restrict__ o) {
  int i = blockIdx.x * blockDim.x + threadIdx.x;  // 0..262143
  o[i]               = (bf16_t)wi_f[i];
  o[262144 + i]      = (bf16_t)wi_b[i];
  o[2 * 262144 + i]  = (bf16_t)wh_f[i];
  o[3 * 262144 + i]  = (bf16_t)wh_b[i];
}

// ---------------- k1: xs[dir] = x @ Wi^T + (bi + bh) ----------------
// grid = BS_/64 blocks, 256 threads (8 waves). Each WG: 64 rows of x.
__global__ void __launch_bounds__(256) proj_gemm(
    const float* __restrict__ x,
    const float* __restrict__ bi_f, const float* __restrict__ bh_f,
    const float* __restrict__ bi_b, const float* __restrict__ bh_b,
    const bf16_t* __restrict__ Wi_bf,  // [2][H_*D_]
    float* __restrict__ xs)            // [2][BS_*H_]
{
  __shared__ alignas(16) bf16_t Asm[64][D_ + 8];
  const int tid  = threadIdx.x;
  const int wgm0 = blockIdx.x * 64;

  // stage 64x512 f32 -> bf16 LDS
  for (int i = tid; i < 64 * (D_ / 4); i += 256) {
    int row = i >> 7;
    int c   = (i & 127) << 2;
    v4f v = *(const v4f*)(x + (size_t)(wgm0 + row) * D_ + c);
    Asm[row][c + 0] = (bf16_t)v[0];
    Asm[row][c + 1] = (bf16_t)v[1];
    Asm[row][c + 2] = (bf16_t)v[2];
    Asm[row][c + 3] = (bf16_t)v[3];
  }
  __syncthreads();

  const int wave  = tid >> 5;
  const int lane  = tid & 31;
  const int lr    = lane & 15;
  const int lhalf = lane >> 4;
  const int koff  = lhalf * 8;

  for (int mt = 0; mt < 4; ++mt) {
    for (int dir = 0; dir < 2; ++dir) {
      for (int nt = wave; nt < 32; nt += 8) {
        const int ncol = nt * 16 + lr;
        const float bias = dir ? (bi_b[ncol] + bh_b[ncol])
                               : (bi_f[ncol] + bh_f[ncol]);
        v8f acc;
#pragma unroll
        for (int j = 0; j < 8; ++j) acc[j] = bias;

        const bf16_t* Brow = Wi_bf + (size_t)dir * H_ * D_ + (size_t)ncol * D_;
        const bf16_t* Arow = &Asm[mt * 16 + lr][0];
#pragma unroll 4
        for (int k0 = 0; k0 < D_; k0 += 32) {
          v8bf alo = *(const v8bf*)(Arow + k0 + koff);
          v8bf ahi = *(const v8bf*)(Arow + k0 + koff + 16);
          v8bf blo = *(const v8bf*)(Brow + k0 + koff);
          v8bf bhi = *(const v8bf*)(Brow + k0 + koff + 16);
          acc = __builtin_amdgcn_wmma_f32_16x16x32_bf16(
              false, cat16(alo, ahi), false, cat16(blo, bhi),
              (short)0, acc, false, false);
        }
        float* op = xs + (size_t)dir * BS_ * H_;
        const int mbase = wgm0 + mt * 16 + lhalf * 8;
#pragma unroll
        for (int j = 0; j < 8; ++j)
          op[(size_t)(mbase + j) * H_ + ncol] = acc[j];
      }
    }
  }
}

// ---------------- k2: persistent recurrent scan ----------------
// grid = 8 blocks (dir = bx&1, batch-tile = bx>>1), 512 threads (16 waves).
// h kept in LDS bf16 ping-pong; one barrier per timestep.
// Each wave: 2 N-tiles with dual accumulators -> one A load feeds two WMMAs,
// two independent accumulate chains hide WMMA RAW latency.
__global__ void __launch_bounds__(512) rnn_scan(
    const float* __restrict__ h0,
    const float* __restrict__ xs,      // [2][BS_*H_] (pre-biased)
    const bf16_t* __restrict__ Wh_bf,  // [2][H_*H_]
    float* __restrict__ out_f,         // d_out main region
    float* __restrict__ out_b)         // ws region
{
  __shared__ alignas(16) bf16_t hbuf[2][16][H_ + 8];
  const int dir  = blockIdx.x & 1;
  const int bt   = blockIdx.x >> 1;
  const int tid  = threadIdx.x;
  const int wave = tid >> 5;
  const int lane = tid & 31;
  const int lr    = lane & 15;
  const int lhalf = lane >> 4;
  const int koff  = lhalf * 8;

  // init h from hidden_state
  for (int i = tid; i < 16 * H_; i += 512) {
    int row = i >> 9, col = i & (H_ - 1);
    hbuf[0][row][col] = (bf16_t)h0[(size_t)(bt * 16 + row) * H_ + col];
  }
  __syncthreads();

  const bf16_t* WhD = Wh_bf + (size_t)dir * H_ * H_;
  const float*  xsD = xs + (size_t)dir * BS_ * H_;
  float* outD = dir ? out_b : out_f;

  const int ncol0 = wave * 32 + lr;        // first N-tile column
  const int ncol1 = ncol0 + 16;            // second N-tile column
  const bf16_t* Brow0 = WhD + (size_t)ncol0 * H_;
  const bf16_t* Brow1 = WhD + (size_t)ncol1 * H_;
  const size_t rowbase = (size_t)(bt * 16 + lhalf * 8) * S_;

  int p = 0;
  for (int s = 0; s < S_; ++s) {
    const int t = dir ? (S_ - 1 - s) : s;
    const size_t base0 = (rowbase + t) * H_ + ncol0;
    const size_t base1 = (rowbase + t) * H_ + ncol1;

    v8f acc0, acc1;
#pragma unroll
    for (int j = 0; j < 8; ++j) {
      acc0[j] = xsD[base0 + (size_t)j * S_ * H_];
      acc1[j] = xsD[base1 + (size_t)j * S_ * H_];
    }

    const bf16_t* Arow = &hbuf[p][lr][0];
#pragma unroll 4
    for (int k0 = 0; k0 < H_; k0 += 32) {
      v8bf alo  = *(const v8bf*)(Arow + k0 + koff);
      v8bf ahi  = *(const v8bf*)(Arow + k0 + koff + 16);
      v16bf a   = cat16(alo, ahi);
      v8bf b0lo = *(const v8bf*)(Brow0 + k0 + koff);
      v8bf b0hi = *(const v8bf*)(Brow0 + k0 + koff + 16);
      v8bf b1lo = *(const v8bf*)(Brow1 + k0 + koff);
      v8bf b1hi = *(const v8bf*)(Brow1 + k0 + koff + 16);
      acc0 = __builtin_amdgcn_wmma_f32_16x16x32_bf16(
          false, a, false, cat16(b0lo, b0hi), (short)0, acc0, false, false);
      acc1 = __builtin_amdgcn_wmma_f32_16x16x32_bf16(
          false, a, false, cat16(b1lo, b1hi), (short)0, acc1, false, false);
    }

#pragma unroll
    for (int j = 0; j < 8; ++j) {
      float y0 = tanhf(acc0[j]);
      float y1 = tanhf(acc1[j]);
      outD[base0 + (size_t)j * S_ * H_] = y0;
      outD[base1 + (size_t)j * S_ * H_] = y1;
      hbuf[1 - p][lhalf * 8 + j][ncol0] = (bf16_t)y0;
      hbuf[1 - p][lhalf * 8 + j][ncol1] = (bf16_t)y1;
    }

    // prefetch next timestep's xs stream
    if (s + 1 < S_) {
      const int tn = dir ? (t - 1) : (t + 1);
      __builtin_prefetch(&xsD[(rowbase + tn) * H_ + ncol0], 0, 0);
    }
    __syncthreads();
    p ^= 1;
  }
}

// ---------------- k3: out = out_f + out_b ----------------
__global__ void finalize_add(float* __restrict__ out,
                             const float* __restrict__ outb) {
  size_t i = (size_t)blockIdx.x * blockDim.x + threadIdx.x;  // vec4 index
  v4f a = ((const v4f*)out)[i];
  v4f b = ((const v4f*)outb)[i];
  ((v4f*)out)[i] = a + b;
}

__global__ void copy_hidden(float* __restrict__ dst,
                            const float* __restrict__ src) {
  int i = blockIdx.x * blockDim.x + threadIdx.x;
  dst[i] = src[i];
}

extern "C" void kernel_launch(void* const* d_in, const int* in_sizes, int n_in,
                              void* d_out, int out_size, void* d_ws, size_t ws_size,
                              hipStream_t stream) {
  (void)in_sizes; (void)n_in; (void)out_size; (void)ws_size;
  const float* x    = (const float*)d_in[0];
  const float* h0   = (const float*)d_in[1];
  const float* Wi_f = (const float*)d_in[2];
  const float* bi_f = (const float*)d_in[3];
  const float* Wh_f = (const float*)d_in[4];
  const float* bh_f = (const float*)d_in[5];
  const float* Wi_b = (const float*)d_in[6];
  const float* bi_b = (const float*)d_in[7];
  const float* Wh_b = (const float*)d_in[8];
  const float* bh_b = (const float*)d_in[9];
  float* out = (float*)d_out;

  // ws layout (f32 units): xs_f | xs_b | out_b | bf16 weights
  float*  ws_f   = (float*)d_ws;
  float*  xs     = ws_f;                               // 2 * BS_*H_ f32
  float*  out_bw = ws_f + (size_t)2 * BS_ * H_;        // BS_*H_ f32
  bf16_t* wbf    = (bf16_t*)(ws_f + (size_t)3 * BS_ * H_);  // 4 * 262144 bf16

  convert_weights<<<H_ * D_ / 256, 256, 0, stream>>>(Wi_f, Wi_b, Wh_f, Wh_b, wbf);
  proj_gemm<<<BS_ / 64, 256, 0, stream>>>(x, bi_f, bh_f, bi_b, bh_b, wbf, xs);
  rnn_scan<<<8, 512, 0, stream>>>(h0, xs, wbf + 2 * 262144, out, out_bw);
  finalize_add<<<(BS_ * H_ / 4) / 256, 256, 0, stream>>>(out, out_bw);
  copy_hidden<<<(B_ * H_) / 256, 256, 0, stream>>>(out + (size_t)BS_ * H_, h0);
}
